// DecodeLayer_55628416417885
// MI455X (gfx1250) — matrix operand order repaired and checked by reading, not compiled, not run on hardware
//
#include <hip/hip_runtime.h>
#include <hip/hip_bf16.h>

typedef unsigned short u16;
typedef unsigned int   u32;
typedef unsigned long long u64;
typedef __bf16 bf16;
typedef __attribute__((ext_vector_type(16))) __bf16 v16bf;
typedef __attribute__((ext_vector_type(8)))  float  v8f;
typedef __attribute__((ext_vector_type(4)))  u32    v4u;
typedef __attribute__((ext_vector_type(8)))  u32    v8u;

union FragU  { u32 u[8]; v16bf v; };
union BfBits { bf16 h; u16 s; };

static __device__ __forceinline__ u16 f2bf_bits(float f) {
    BfBits t; t.h = (bf16)f; return t.s;
}
// LDS byte offset of a generic __shared__ pointer (generic LDS addr[31:0] == offset)
static __device__ __forceinline__ u32 lds_off(const void* p) {
    return (u32)(size_t)p;
}
// CDNA5 async global->LDS, 16B per lane, GVS addressing (uniform 64b base + 32b voffset)
static __device__ __forceinline__ void async_copy_b128(u32 lds_addr, u32 voff, u64 sbase) {
    asm volatile("global_load_async_to_lds_b128 %0, %1, %2"
                 :: "v"(lds_addr), "v"(voff), "s"(sbase) : "memory");
}
static __device__ __forceinline__ void wait_async0() {
    asm volatile("s_wait_asynccnt 0x0" ::: "memory");
}
static __device__ __forceinline__ void wait_tensor0() {
#if __has_builtin(__builtin_amdgcn_s_wait_tensorcnt)
    __builtin_amdgcn_s_wait_tensorcnt(0);
#else
    asm volatile("s_wait_tensorcnt 0x0" ::: "memory");
#endif
}

// ---------------------------------------------------------------------------
// TDM: DMA one 32(k) x 128(n) bf16 tile (row stride N elems) global -> LDS,
// with 16B pad after each 256B row => LDS row stride 136 halves.
// D# group0: count=1 | lds_addr | global_addr[56:0] | type=2
// D# group1: data_size=2B, pad_enable, pad_interval=256B, pad_amount=16B,
//            tensor_dim 128x32, tile_dim 128x32, dim0_stride=N
// ---------------------------------------------------------------------------
static __device__ __forceinline__ void tdm_load_tile_b(u32 lds_addr, u64 gaddr, u32 N) {
    v4u g0;
    g0[0] = 1u;                                             // count = 1 (valid)
    g0[1] = lds_addr;                                       // LDS byte address
    g0[2] = (u32)gaddr;                                     // global_addr[31:0]
    g0[3] = (u32)((gaddr >> 32) & 0x01FFFFFFull)            // global_addr[56:32]
          | 0x80000000u;                                    // type = 2 ("image")
    v8u g1;
    g1[0] = 0x07510000u;    // data_size=1(2B)<<16 | pad_en<<20 | intv=5<<22 | amt=3<<25
    g1[1] = (128u & 0xFFFFu) << 16;                         // tensor_dim0 = 128 (lo16)
    g1[2] = (32u  & 0xFFFFu) << 16;                         // tensor_dim1 = 32  (lo16)
    g1[3] = (128u << 16);                                   // tile_dim0 = 128
    g1[4] = 32u;                                            // tile_dim1 = 32, tile_dim2 = 0
    g1[5] = N;                                              // tensor_dim0_stride[31:0]
    g1[6] = 0u;
    g1[7] = 0u;
    asm volatile("tensor_load_to_lds %0, %1, null, null" :: "s"(g0), "s"(g1) : "memory");
}

// ---------------------------------------------------------------------------
// fp32 -> bf16 elementwise (weight conversion)
// ---------------------------------------------------------------------------
__global__ void cvt_f32_bf16(const float* __restrict__ in, bf16* __restrict__ out, int n) {
    for (int i = blockIdx.x * blockDim.x + threadIdx.x; i < n; i += gridDim.x * blockDim.x)
        out[i] = (bf16)in[i];
}

// ---------------------------------------------------------------------------
// LayerNorm over 1024 cols, one block per row, fp32 in -> bf16 out
// ---------------------------------------------------------------------------
__global__ __launch_bounds__(256) void ln_bf16_kernel(
    const float* __restrict__ x, const float* __restrict__ g,
    const float* __restrict__ be, bf16* __restrict__ out) {
    int row = blockIdx.x;
    int t   = threadIdx.x;
    const float4 xv = *(const float4*)(x + (size_t)row * 1024 + t * 4);
    float xa[4] = {xv.x, xv.y, xv.z, xv.w};
    float s  = xa[0] + xa[1] + xa[2] + xa[3];
    float s2 = xa[0]*xa[0] + xa[1]*xa[1] + xa[2]*xa[2] + xa[3]*xa[3];
#pragma unroll
    for (int off = 16; off >= 1; off >>= 1) {
        s  += __shfl_xor(s,  off, 32);
        s2 += __shfl_xor(s2, off, 32);
    }
    __shared__ float red[16];
    int w = t >> 5, lane = t & 31;
    if (lane == 0) { red[w] = s; red[8 + w] = s2; }
    __syncthreads();
    if (t == 0) {
        float a = 0.f, b2 = 0.f;
        for (int i = 0; i < 8; ++i) { a += red[i]; b2 += red[8 + i]; }
        red[0] = a; red[8] = b2;
    }
    __syncthreads();
    float mu   = red[0] * (1.0f / 1024.0f);
    float var  = red[8] * (1.0f / 1024.0f) - mu * mu;
    float rstd = rsqrtf(var + 1e-5f);
#pragma unroll
    for (int j = 0; j < 4; ++j) {
        int c = t * 4 + j;
        out[(size_t)row * 1024 + c] = (bf16)((xa[j] - mu) * rstd * g[c] + be[c]);
    }
}

// ---------------------------------------------------------------------------
// bf16 GEMM, f32 accumulate via WMMA.
//   A tile: double-buffered async global->LDS (ASYNCcnt)
//   B tile: double-buffered TDM tensor_load_to_lds (TENSORcnt), wave 0 issues
//   EPI=0: bf16 out (bias)          EPI=1: bf16 out (bias + exact GELU)
//   EPI=2: f32 out (bias + residual add)
// Block tile 128x128, BK=32, 256 threads = 8 waves, each wave 64x32 (4x2 tiles)
// ---------------------------------------------------------------------------
template <int EPI>
__global__ __launch_bounds__(256) void gemm_bf16_kernel(
    const bf16* __restrict__ A, const bf16* __restrict__ B,
    const float* __restrict__ bias, const float* __restrict__ res,
    bf16* __restrict__ Cb, float* __restrict__ Cf,
    int M, int N, int K) {
    __shared__ u16 As[2][128 * 40];   // A tile [m][k], stride 40 (16B-aligned rows)
    __shared__ u16 Bs[2][32 * 136];   // B tile [k][n], stride 136 (TDM pad = 16B/row)

    const int m0 = blockIdx.y * 128;
    const int n0 = blockIdx.x * 128;
    const int t  = threadIdx.x;
    const int lane = t & 31, w = t >> 5;
    const int wm = w & 1, wn = w >> 1;          // 2 x 4 wave grid
    const int li = lane & 15, hi = lane >> 4;

    // --- A staging map: 16B (8 bf16) per lane per issue; 2 issues per tile
    const int iA = t * 8;
    const int rA = iA >> 5, cA = iA & 31;            // rep1: row += 64
    const u32 offA = (u32)(((m0 + rA) * K + cA) * 2);
    const u64 baseA = (u64)(size_t)A;
    const u64 baseB = (u64)(size_t)B;

    v8f acc[4][2] = {};
    const int iters = K >> 5;

    // prologue: stage k0 = 0 into buffer 0
    async_copy_b128(lds_off(&As[0][rA * 40 + cA]),        offA,                baseA);
    async_copy_b128(lds_off(&As[0][(rA + 64) * 40 + cA]), offA + (u32)(128*K), baseA);
    if (w == 0)
        tdm_load_tile_b(lds_off(&Bs[0][0]), baseB + (u64)(n0 * 2), (u32)N);

    for (int it = 0; it < iters; ++it) {
        wait_async0();        // own async A writes to buf[it&1] complete
        wait_tensor0();       // wave0: TDM B tile complete (no-op elsewhere)
        __syncthreads();      // publish tiles; prior reads retired
        if (it + 1 < iters) {
            const int nb = (it + 1) & 1;
            const u32 ka = offA + (u32)((it + 1) * 64);          // += 32 (k) * 2B
            async_copy_b128(lds_off(&As[nb][rA * 40 + cA]),        ka,                baseA);
            async_copy_b128(lds_off(&As[nb][(rA + 64) * 40 + cA]), ka + (u32)(128*K), baseA);
            if (w == 0)
                tdm_load_tile_b(lds_off(&Bs[nb][0]),
                                baseB + (u64)(((size_t)(it + 1) * 32 * N + n0) * 2),
                                (u32)N);
        }
        const u16* Ab = As[it & 1];
        const u16* Bb = Bs[it & 1];

        // A fragments: documented wave32 16-bit A layout, ushort2-packed
        FragU af[4], bfr[2];
#pragma unroll
        for (int mt = 0; mt < 4; ++mt) {
            int row = wm * 64 + mt * 16 + li;
#pragma unroll
            for (int p = 0; p < 8; ++p) {
                int kk = ((p >> 2) << 4) + (hi << 3) + ((p & 3) << 1);
                af[mt].u[p] = *(const u32*)&Ab[row * 40 + kk];
            }
        }
        // B fragments: from row-major [k][n] tile, per-half ds loads
#pragma unroll
        for (int nt = 0; nt < 2; ++nt) {
            int col = wn * 32 + nt * 16 + li;
#pragma unroll
            for (int p = 0; p < 8; ++p) {
                int k0b = ((p >> 2) << 4) + (hi << 3) + ((p & 3) << 1);
                u16 h0 = Bb[k0b * 136 + col];
                u16 h1 = Bb[(k0b + 1) * 136 + col];
                bfr[nt].u[p] = (u32)h0 | ((u32)h1 << 16);
            }
        }
#pragma unroll
        for (int mt = 0; mt < 4; ++mt)
#pragma unroll
            for (int nt = 0; nt < 2; ++nt)
                acc[mt][nt] = __builtin_amdgcn_wmma_f32_16x16x32_bf16(
                    false, af[mt].v, false, bfr[nt].v, (short)0, acc[mt][nt],
                    false, false);
    }

    // -- epilogue
#pragma unroll
    for (int mt = 0; mt < 4; ++mt)
#pragma unroll
        for (int nt = 0; nt < 2; ++nt)
#pragma unroll
            for (int r = 0; r < 8; ++r) {
                int gm = m0 + wm * 64 + mt * 16 + r + (hi << 3);
                int gn = n0 + wn * 32 + nt * 16 + li;
                float v = acc[mt][nt][r] + bias[gn];
                if (EPI == 1)
                    v = 0.5f * v * (1.0f + erff(v * 0.70710678118654752f));
                if (EPI == 2)
                    Cf[(size_t)gm * N + gn] = v + res[(size_t)gm * N + gn];
                else
                    Cb[(size_t)gm * N + gn] = (bf16)v;
            }
}

// ---------------------------------------------------------------------------
// Flash attention: one wave per 16-query tile. qkv is [B*S, 3*E] bf16 with
// column layout [which(3)][head(16)][d(64)]. out is [B*S, E] bf16.
// Per 32-key block: 4 score WMMAs + online softmax + 4 P@V WMMAs.
// ---------------------------------------------------------------------------
__global__ __launch_bounds__(128) void attn_kernel(const bf16* __restrict__ qkv,
                                                   bf16* __restrict__ out) {
    __shared__ u16 Pld[4][16 * 32];   // per-wave P scratch (C-layout -> A-layout)

    const int t = threadIdx.x;
    const int w = t >> 5, lane = t & 31;
    const int li = lane & 15, hi = lane >> 4;
    const int gw = blockIdx.x * 4 + w;        // 0 .. 8191
    const int b  = gw >> 11;                  // / 2048
    const int r1 = gw & 2047;
    const int h  = r1 >> 7;
    const int q0 = (r1 & 127) << 4;
    const size_t bt0 = (size_t)b * 2048;
    const size_t ld  = 3072;
    const int hoff   = h * 64;

    // Q fragments (A-layout), d = 0..31 and 32..63
    FragU aq[2];
#pragma unroll
    for (int kc = 0; kc < 2; ++kc)
#pragma unroll
        for (int p = 0; p < 8; ++p) {
            int d = kc * 32 + ((p >> 2) << 4) + (hi << 3) + ((p & 3) << 1);
            aq[kc].u[p] = *(const u32*)(qkv + (bt0 + q0 + li) * ld + hoff + d);
        }

    v8f accO[4] = {};
    float mrow[8], lrow[8];
#pragma unroll
    for (int r = 0; r < 8; ++r) { mrow[r] = -1e30f; lrow[r] = 0.f; }

    for (int j0 = 0; j0 <= q0 + 15; j0 += 32) {
        // prefetch next key block's K rows (global_prefetch_b8)
        if (j0 + 32 <= q0 + 15) {
            __builtin_prefetch(qkv + (bt0 + j0 + 32 + li) * ld + 1024 + hoff, 0, 1);
            __builtin_prefetch(qkv + (bt0 + j0 + 48 + li) * ld + 1024 + hoff, 0, 1);
        }
        // ---- scores S[16 x 32] in two 16x16 f32 tiles
        v8f s[2] = {};
#pragma unroll
        for (int c = 0; c < 2; ++c) {
#pragma unroll
            for (int kc = 0; kc < 2; ++kc) {
                FragU bk;
                int key = j0 + c * 16 + li;
#pragma unroll
                for (int p = 0; p < 8; ++p) {
                    int d = kc * 32 + ((p >> 2) << 4) + (hi << 3) + ((p & 3) << 1);
                    bk.u[p] = *(const u32*)(qkv + (bt0 + key) * ld + 1024 + hoff + d);
                }
                s[c] = __builtin_amdgcn_wmma_f32_16x16x32_bf16(
                    false, aq[kc].v, false, bk.v, (short)0, s[c], false, false);
            }
        }
        // ---- causal mask + online softmax (rows grouped by half-wave)
#pragma unroll
        for (int r = 0; r < 8; ++r) {
            int qrow = q0 + r + (hi << 3);
            float v0 = s[0][r] * 0.125f;
            float v1 = s[1][r] * 0.125f;
            if (j0 + li > qrow)      v0 = -1e30f;
            if (j0 + 16 + li > qrow) v1 = -1e30f;
            float mx = fmaxf(v0, v1);
#pragma unroll
            for (int off = 1; off < 16; off <<= 1)
                mx = fmaxf(mx, __shfl_xor(mx, off, 16));
            float mnew = fmaxf(mrow[r], mx);
            float p0 = __expf(v0 - mnew);
            float p1 = __expf(v1 - mnew);
            float rs = p0 + p1;
#pragma unroll
            for (int off = 1; off < 16; off <<= 1)
                rs += __shfl_xor(rs, off, 16);
            float corr = __expf(mrow[r] - mnew);
            lrow[r] = lrow[r] * corr + rs;
            mrow[r] = mnew;
            s[0][r] = p0; s[1][r] = p1;
#pragma unroll
            for (int dt = 0; dt < 4; ++dt) accO[dt][r] *= corr;
        }
        // ---- P: C-layout -> bf16 row-major [16][32] in per-wave LDS
#pragma unroll
        for (int c = 0; c < 2; ++c)
#pragma unroll
            for (int r = 0; r < 8; ++r) {
                int m = r + (hi << 3);
                Pld[w][m * 32 + c * 16 + li] = f2bf_bits(s[c][r]);
            }
        asm volatile("s_wait_dscnt 0x0" ::: "memory");  // intra-wave LDS RAW
        // ---- reload P as A fragment
        FragU ap;
#pragma unroll
        for (int p = 0; p < 8; ++p) {
            int kk = ((p >> 2) << 4) + (hi << 3) + ((p & 3) << 1);
            ap.u[p] = *(const u32*)&Pld[w][li * 32 + kk];
        }
        // ---- O += P @ V  (four d-chunks of 16)
#pragma unroll
        for (int dt = 0; dt < 4; ++dt) {
            FragU bv;
#pragma unroll
            for (int p = 0; p < 8; ++p) {
                int k0b = j0 + ((p >> 2) << 4) + (hi << 3) + ((p & 3) << 1);
                u16 h0 = *(const u16*)(qkv + (bt0 + k0b)     * ld + 2048 + hoff + dt * 16 + li);
                u16 h1 = *(const u16*)(qkv + (bt0 + k0b + 1) * ld + 2048 + hoff + dt * 16 + li);
                bv.u[p] = (u32)h0 | ((u32)h1 << 16);
            }
            accO[dt] = __builtin_amdgcn_wmma_f32_16x16x32_bf16(
                false, ap.v, false, bv.v, (short)0, accO[dt], false, false);
        }
    }

    // ---- normalize and write out [token, h*64 + d] as bf16
#pragma unroll
    for (int r = 0; r < 8; ++r) {
        float inv = 1.0f / lrow[r];
        size_t rowi = (bt0 + q0 + r + (hi << 3)) * 1024;
#pragma unroll
        for (int dt = 0; dt < 4; ++dt)
            out[rowi + hoff + dt * 16 + li] = (bf16)(accO[dt][r] * inv);
    }
}

// ---------------------------------------------------------------------------
// Launch: full pre-LN transformer block  (B=4, S=2048, E=1024, H=16)
// ---------------------------------------------------------------------------
extern "C" void kernel_launch(void* const* d_in, const int* in_sizes, int n_in,
                              void* d_out, int out_size, void* d_ws, size_t ws_size,
                              hipStream_t stream) {
    (void)in_sizes; (void)n_in; (void)out_size; (void)ws_size;

    const float* x      = (const float*)d_in[0];
    const float* ln1_g  = (const float*)d_in[1];
    const float* ln1_b  = (const float*)d_in[2];
    const float* w_attn = (const float*)d_in[3];
    const float* b_attn = (const float*)d_in[4];
    const float* w_proj = (const float*)d_in[5];
    const float* b_proj = (const float*)d_in[6];
    const float* ln2_g  = (const float*)d_in[7];
    const float* ln2_b  = (const float*)d_in[8];
    const float* w_fc   = (const float*)d_in[9];
    const float* b_fc   = (const float*)d_in[10];
    const float* w_fc2  = (const float*)d_in[11];
    const float* b_fc2  = (const float*)d_in[12];

    char* w = (char*)d_ws;                       // workspace layout (bytes)
    bf16*  wb_attn = (bf16*)(w + 0);             //  6.0 MB  1024x3072 bf16
    bf16*  wb_proj = (bf16*)(w + 6291456);       //  2.0 MB  1024x1024
    bf16*  wb_fc   = (bf16*)(w + 8388608);       //  8.0 MB  1024x4096
    bf16*  wb_fc2  = (bf16*)(w + 16777216);      //  8.0 MB  4096x1024
    bf16*  xn      = (bf16*)(w + 25165824);      // 16.0 MB  8192x1024 (reused)
    bf16*  qkv     = (bf16*)(w + 41943040);      // 48.0 MB  8192x3072
    bf16*  attn_o  = (bf16*)(w + 92274688);      // 16.0 MB  8192x1024
    float* xmid    = (float*)(w + 109051904);    // 32.0 MB  8192x1024 f32
    bf16*  hbuf    = (bf16*)(w + 142606336);     // 64.0 MB  8192x4096

    // 1) weights -> bf16
    cvt_f32_bf16<<<2048, 256, 0, stream>>>(w_attn, wb_attn, 1024 * 3072);
    cvt_f32_bf16<<<1024, 256, 0, stream>>>(w_proj, wb_proj, 1024 * 1024);
    cvt_f32_bf16<<<2048, 256, 0, stream>>>(w_fc,   wb_fc,   1024 * 4096);
    cvt_f32_bf16<<<2048, 256, 0, stream>>>(w_fc2,  wb_fc2,  4096 * 1024);

    // 2) LN1
    ln_bf16_kernel<<<8192, 256, 0, stream>>>(x, ln1_g, ln1_b, xn);

    // 3) QKV = xn @ w_attn + b_attn      [8192,3072]
    gemm_bf16_kernel<0><<<dim3(3072 / 128, 8192 / 128), 256, 0, stream>>>(
        xn, wb_attn, b_attn, nullptr, qkv, nullptr, 8192, 3072, 1024);

    // 4) causal flash attention          [8192,1024]
    attn_kernel<<<2048, 128, 0, stream>>>(qkv, attn_o);

    // 5) x_mid = x + attn_o @ w_proj + b_proj   (f32)
    gemm_bf16_kernel<2><<<dim3(1024 / 128, 8192 / 128), 256, 0, stream>>>(
        attn_o, wb_proj, b_proj, x, nullptr, xmid, 8192, 1024, 1024);

    // 6) LN2
    ln_bf16_kernel<<<8192, 256, 0, stream>>>(xmid, ln2_g, ln2_b, xn);

    // 7) h = gelu(xn @ w_fc + b_fc)      [8192,4096]
    gemm_bf16_kernel<1><<<dim3(4096 / 128, 8192 / 128), 256, 0, stream>>>(
        xn, wb_fc, b_fc, nullptr, hbuf, nullptr, 8192, 4096, 1024);

    // 8) out = x_mid + h @ w_fc2 + b_fc2  (f32)
    gemm_bf16_kernel<2><<<dim3(1024 / 128, 8192 / 128), 256, 0, stream>>>(
        hbuf, wb_fc2, b_fc2, xmid, nullptr, (float*)d_out, 8192, 1024, 4096);
}
